// HeteroRGCN_13280038879654
// MI455X (gfx1250) — compile-verified, matrix operand order checked by reading, not run on hardware
//
#include <hip/hip_runtime.h>
#include <hip/hip_bf16.h>

typedef __attribute__((ext_vector_type(16))) __bf16 v16bf;
typedef __attribute__((ext_vector_type(8)))  float  v8f;

#define NNODES 100000
#define NREL   4
#define NEDGE  200000
#define DIN    128
#define DHID   256

#define GMODE_BF16 0
#define GMODE_F32  1
#define GMODE_ACC  2

// ---------------- small utility kernels ----------------

// out[n*K + k] = (bf16) in[k*N + n]   (N = DHID), i.e. store W^T for coalesced GEMM staging
__global__ void cvt_transpose_kernel(const float* __restrict__ in,
                                     __bf16* __restrict__ out, int K) {
  int i = blockIdx.x * 256 + threadIdx.x;   // i = n*K + k
  if (i < K * DHID) {
    int n = i / K;
    int k = i - n * K;
    out[i] = (__bf16)in[(size_t)k * DHID + n];
  }
}

__global__ void zero_f32_kernel(float* __restrict__ p, long long n) {
  long long i = ((long long)blockIdx.x * 256 + threadIdx.x) * 4;
  if (i + 3 < n) {
    *(float4*)(p + i) = make_float4(0.f, 0.f, 0.f, 0.f);
  } else {
    for (long long j = i; j < n; ++j) p[j] = 0.f;
  }
}

// one pass over all R*E edges; counts out-degree (src) and in-degree (dst)
__global__ void deg_kernel(const int* __restrict__ src, const int* __restrict__ dst,
                           float* __restrict__ ns, float* __restrict__ nd) {
  long long i = (long long)blockIdx.x * 256 + threadIdx.x;
  if (i < (long long)NREL * NEDGE) {
    int r = (int)(i / NEDGE);
    unsafeAtomicAdd(ns + (size_t)r * NNODES + src[i], 1.0f);
    unsafeAtomicAdd(nd + (size_t)r * NNODES + dst[i], 1.0f);
  }
}

__global__ void norm_kernel(float* __restrict__ p, int n) {
  int i = blockIdx.x * 256 + threadIdx.x;
  if (i < n) {
    float v = p[i];
    p[i] = rsqrtf(v < 1.0f ? 1.0f : v);
  }
}

// ---------------- WMMA bf16 GEMM ----------------
// out[M,256] = op( (A[M,K] * rowScale) @ B[K,256] + bias )
// B supplied pre-transposed: Bt[n*K + k].
// Block: 256 threads = 8 waves. Block tile 256(M) x 64(N), K-step 32.
// Wave w owns rows [32w, 32w+32): two A fragments, 4 col-groups -> 8 WMMA/k-step.

template <int MODE, bool SCALED>
__global__ __launch_bounds__(256) void gemm_bf16_kernel(
    const float* __restrict__ A, const float* __restrict__ rowScale,
    const __bf16* __restrict__ Bt, const float* __restrict__ bias,
    float* __restrict__ outF, __bf16* __restrict__ outH,
    int M, int K, float alpha) {
  __shared__ __bf16 As[256][36];   // [row][k], 72B row stride (8B aligned)
  __shared__ __bf16 Bs[64][40];    // [col][k], 80B row stride (16B aligned)

  const int t    = threadIdx.x;
  const int lane = t & 31;
  const int wave = t >> 5;
  const int half = lane >> 4;      // 0: lanes 0-15, 1: lanes 16-31
  const int rl   = lane & 15;
  const int rowBase = blockIdx.x * 256;
  const int colBase = blockIdx.y * 64;

  v8f acc0[4] = {};
  v8f acc1[4] = {};

  for (int kb = 0; kb < K; kb += 32) {
    // stage A tile: 256x32 fp32 -> bf16, branchless (clamped row + zeroed scale for OOB)
    #pragma unroll
    for (int i = 0; i < 8; ++i) {
      int fi = i * 256 + t;          // float4 index within tile
      int ar = fi >> 3;              // row in tile
      int ac = (fi & 7) << 2;        // col in tile
      int grow = rowBase + ar;
      int crow = grow < M ? grow : M - 1;
      float scl;
      if (SCALED) {
        float rs = rowScale[crow];
        scl = (grow < M) ? rs : 0.0f;
      } else {
        scl = (grow < M) ? 1.0f : 0.0f;
      }
      float4 v = *(const float4*)(A + (size_t)crow * K + kb + ac);
      union { uint2 u2; __bf16 h[4]; } pk;
      pk.h[0] = (__bf16)(v.x * scl);
      pk.h[1] = (__bf16)(v.y * scl);
      pk.h[2] = (__bf16)(v.z * scl);
      pk.h[3] = (__bf16)(v.w * scl);
      *(uint2*)&As[ar][ac] = pk.u2;
    }
    // stage B tile: 64 cols x 32 k, already transposed in global -> 16B copies
    {
      int col = t >> 2;              // 64 cols
      int k0  = (t & 3) << 3;        // 0,8,16,24
      uint4 bv = *(const uint4*)(Bt + (size_t)(colBase + col) * K + kb + k0);
      *(uint4*)&Bs[col][k0] = bv;
    }
    __syncthreads();

    // A fragments per ISA 16-bit A 16x32 layout
    v16bf af0, af1;
    #pragma unroll
    for (int g = 0; g < 8; ++g) {
      int k = ((g < 4) ? (2 * g) : (16 + 2 * (g - 4))) + half * 8;
      af0[2 * g]     = As[wave * 32 + rl][k];
      af0[2 * g + 1] = As[wave * 32 + rl][k + 1];
      af1[2 * g]     = As[wave * 32 + 16 + rl][k];
      af1[2 * g + 1] = As[wave * 32 + 16 + rl][k + 1];
    }
    #pragma unroll
    for (int nt = 0; nt < 4; ++nt) {
      v16bf bfr;
      #pragma unroll
      for (int g = 0; g < 8; ++g) {
        int k = ((g < 4) ? (2 * g) : (16 + 2 * (g - 4))) + half * 8;
        bfr[2 * g]     = Bs[nt * 16 + rl][k];
        bfr[2 * g + 1] = Bs[nt * 16 + rl][k + 1];
      }
      acc0[nt] = __builtin_amdgcn_wmma_f32_16x16x32_bf16(
          false, af0, false, bfr, (short)0, acc0[nt], false, false);
      acc1[nt] = __builtin_amdgcn_wmma_f32_16x16x32_bf16(
          false, af1, false, bfr, (short)0, acc1[nt], false, false);
    }
    __syncthreads();
  }

  // epilogue: D layout (f32): VGPR v -> M = v + 8*half, N = rl
  #pragma unroll
  for (int nt = 0; nt < 4; ++nt) {
    int col = colBase + nt * 16 + rl;
    float bv = bias ? bias[col] : 0.0f;
    #pragma unroll
    for (int f = 0; f < 2; ++f) {
      #pragma unroll
      for (int v = 0; v < 8; ++v) {
        int row = rowBase + wave * 32 + f * 16 + half * 8 + v;
        if (row < M) {
          float val = (f ? acc1[nt][v] : acc0[nt][v]) + bv;
          size_t idx = (size_t)row * DHID + col;
          if (MODE == GMODE_BF16)      outH[idx] = (__bf16)val;
          else if (MODE == GMODE_F32)  outF[idx] = alpha * val;
          else                         outF[idx] += alpha * val;
        }
      }
    }
  }
}

// ---------------- gather * ns * ew -> atomic scatter-sum to agg ----------------
// 32 lanes per edge, 8 contiguous features each (DHID=256); 8 edges per block.

__global__ void scatter_kernel(const float* __restrict__ x, const __bf16* __restrict__ ew,
                               const int* __restrict__ src, const int* __restrict__ dst,
                               const float* __restrict__ ns, float* __restrict__ agg, int nE) {
  int t = threadIdx.x;
  int e = blockIdx.x * 8 + (t >> 5);
  if (e >= nE) return;
  int h = (t & 31) << 3;
  int s = src[e];
  int d = dst[e];
  float c = ns[s];
  float4 xv0 = *(const float4*)(x + (size_t)s * DHID + h);
  float4 xv1 = *(const float4*)(x + (size_t)s * DHID + h + 4);
  union { uint4 u4; __bf16 hh[8]; } uw;
  uw.u4 = *(const uint4*)(ew + (size_t)e * DHID + h);
  float* ap = agg + (size_t)d * DHID + h;
  unsafeAtomicAdd(ap + 0, xv0.x * c * (float)uw.hh[0]);
  unsafeAtomicAdd(ap + 1, xv0.y * c * (float)uw.hh[1]);
  unsafeAtomicAdd(ap + 2, xv0.z * c * (float)uw.hh[2]);
  unsafeAtomicAdd(ap + 3, xv0.w * c * (float)uw.hh[3]);
  unsafeAtomicAdd(ap + 4, xv1.x * c * (float)uw.hh[4]);
  unsafeAtomicAdd(ap + 5, xv1.y * c * (float)uw.hh[5]);
  unsafeAtomicAdd(ap + 6, xv1.z * c * (float)uw.hh[6]);
  unsafeAtomicAdd(ap + 7, xv1.w * c * (float)uw.hh[7]);
}

// ---------------- host orchestration ----------------

static inline int cdiv(long long a, long long b) { return (int)((a + b - 1) / b); }

extern "C" void kernel_launch(void* const* d_in, const int* in_sizes, int n_in,
                              void* d_out, int out_size, void* d_ws, size_t ws_size,
                              hipStream_t stream) {
  (void)in_sizes; (void)n_in; (void)out_size; (void)ws_size;

  const float* node_feat  = (const float*)d_in[0];
  const float* edge_w_raw = (const float*)d_in[1];
  const int*   src        = (const int*)d_in[2];
  const int*   dst        = (const int*)d_in[3];
  const float* Wn         = (const float*)d_in[4];
  const float* bn         = (const float*)d_in[5];
  const float* Wr         = (const float*)d_in[6];
  const float* br         = (const float*)d_in[7];
  const float* W1         = (const float*)d_in[8];
  const float* b1         = (const float*)d_in[9];
  const float* W2         = (const float*)d_in[10];
  const float* b2         = (const float*)d_in[11];
  float* out = (float*)d_out;

  // workspace layout (256B aligned slabs)
  char* base = (char*)d_ws;
  size_t off = 0;
  auto take = [&](size_t bytes) -> char* {
    char* r = base + off;
    off += (bytes + 255) & ~(size_t)255;
    return r;
  };
  __bf16* WrT  = (__bf16*)take((size_t)DIN * DHID * 2);          // [DHID][DIN]
  __bf16* WnT  = (__bf16*)take((size_t)DIN * DHID * 2);          // [DHID][DIN]
  __bf16* W1T  = (__bf16*)take((size_t)NREL * DHID * DHID * 2);  // R x [DHID][DHID]
  __bf16* W2T  = (__bf16*)take((size_t)NREL * DHID * DHID * 2);
  float*  ns   = (float*)take((size_t)NREL * NNODES * 4);
  float*  nd   = (float*)take((size_t)NREL * NNODES * 4);
  float*  x0   = (float*)take((size_t)NNODES * DHID * 4);
  float*  h1   = (float*)take((size_t)NNODES * DHID * 4);
  float*  agg  = (float*)take((size_t)NNODES * DHID * 4);
  __bf16* ewh  = (__bf16*)take((size_t)NREL * NEDGE * DHID * 2);

  // 1) convert + transpose weights to bf16
  cvt_transpose_kernel<<<cdiv(DIN * DHID, 256), 256, 0, stream>>>(Wr, WrT, DIN);
  cvt_transpose_kernel<<<cdiv(DIN * DHID, 256), 256, 0, stream>>>(Wn, WnT, DIN);
  for (int r = 0; r < NREL; ++r) {
    cvt_transpose_kernel<<<cdiv(DHID * DHID, 256), 256, 0, stream>>>(
        W1 + (size_t)r * DHID * DHID, W1T + (size_t)r * DHID * DHID, DHID);
    cvt_transpose_kernel<<<cdiv(DHID * DHID, 256), 256, 0, stream>>>(
        W2 + (size_t)r * DHID * DHID, W2T + (size_t)r * DHID * DHID, DHID);
  }

  // 2) degrees -> normalization factors (ns and nd contiguous)
  zero_f32_kernel<<<cdiv((long long)2 * NREL * NNODES, 1024), 256, 0, stream>>>(
      ns, (long long)2 * NREL * NNODES);
  deg_kernel<<<cdiv((long long)NREL * NEDGE, 256), 256, 0, stream>>>(src, dst, ns, nd);
  norm_kernel<<<cdiv(2 * NREL * NNODES, 256), 256, 0, stream>>>(ns, 2 * NREL * NNODES);

  // 3) ew = edge_w_raw @ Wr + br, stored bf16  (M = R*E = 800000, one big GEMM)
  {
    dim3 grid(cdiv((long long)NREL * NEDGE, 256), DHID / 64);
    gemm_bf16_kernel<GMODE_BF16, false><<<grid, 256, 0, stream>>>(
        edge_w_raw, nullptr, WrT, br, nullptr, ewh, NREL * NEDGE, DIN, 1.0f);
  }

  // 4) x0 = node_feat @ Wn + bn (fp32)
  {
    dim3 grid(cdiv(NNODES, 256), DHID / 64);
    gemm_bf16_kernel<GMODE_F32, false><<<grid, 256, 0, stream>>>(
        node_feat, nullptr, WnT, bn, x0, nullptr, NNODES, DIN, 1.0f);
  }

  // 5) two hetero-conv layers
  const float*  xin  = x0;
  float*        hout = h1;
  const __bf16* Wl   = W1T;
  const float*  bl   = b1;
  for (int layer = 0; layer < 2; ++layer) {
    for (int r = 0; r < NREL; ++r) {
      zero_f32_kernel<<<cdiv((long long)NNODES * DHID, 1024), 256, 0, stream>>>(
          agg, (long long)NNODES * DHID);
      scatter_kernel<<<cdiv(NEDGE, 8), 256, 0, stream>>>(
          xin, ewh + (size_t)r * NEDGE * DHID,
          src + (size_t)r * NEDGE, dst + (size_t)r * NEDGE,
          ns + (size_t)r * NNODES, agg, NEDGE);
      dim3 grid(cdiv(NNODES, 256), DHID / 64);
      if (r == 0) {
        gemm_bf16_kernel<GMODE_F32, true><<<grid, 256, 0, stream>>>(
            agg, nd + (size_t)r * NNODES,
            Wl + (size_t)r * DHID * DHID, bl + (size_t)r * DHID,
            hout, nullptr, NNODES, DHID, 1.0f / NREL);
      } else {
        gemm_bf16_kernel<GMODE_ACC, true><<<grid, 256, 0, stream>>>(
            agg, nd + (size_t)r * NNODES,
            Wl + (size_t)r * DHID * DHID, bl + (size_t)r * DHID,
            hout, nullptr, NNODES, DHID, 1.0f / NREL);
      }
    }
    xin = h1; hout = out; Wl = W2T; bl = b2;
  }
}
